// PointStateMixer_55868934586532
// MI455X (gfx1250) — matrix operand order repaired and checked by reading, not compiled, not run on hardware
//
#include <hip/hip_runtime.h>
#include <hip/hip_bf16.h>
#include <cstdint>

#define CCH 512
#define HH  1024
#define KK  5
#define PADK 2
#define EPSF 1e-5f

// CDNA5 Tensor Data Mover path (6-arg clang-23 builtin, confirmed this toolchain).
#define USE_TDM_TRY 1

#if defined(__has_builtin)
#if USE_TDM_TRY && __has_builtin(__builtin_amdgcn_tensor_load_to_lds) && \
    __has_builtin(__builtin_amdgcn_s_wait_tensorcnt)
#define USE_TDM 1
#endif
#endif
#ifndef USE_TDM
#define USE_TDM 0
#endif

typedef __attribute__((ext_vector_type(16))) __bf16   v16bf;
typedef __attribute__((ext_vector_type(8)))  __bf16   v8bf;
typedef __attribute__((ext_vector_type(8)))  float    v8f;
typedef __attribute__((ext_vector_type(4)))  unsigned v4u;
typedef __attribute__((ext_vector_type(8)))  int      v8i_;
typedef __attribute__((ext_vector_type(4)))  int      v4i_;

// ---------- bf16 bit helpers (RNE), no arithmetic on __bf16 ----------
__device__ __forceinline__ unsigned bfbits(float f) {
  unsigned u; __builtin_memcpy(&u, &f, 4);
  return (u + 0x7FFFu + ((u >> 16) & 1u)) >> 16;
}
__device__ __forceinline__ unsigned pkbf(float lo, float hi) {
  return bfbits(lo) | (bfbits(hi) << 16);
}
__device__ __forceinline__ __bf16 f2bf(float f) {
  unsigned short h = (unsigned short)bfbits(f);
  __bf16 o; __builtin_memcpy(&o, &h, 2); return o;
}
__device__ __forceinline__ float bf2f(__bf16 b) {
  unsigned short h; __builtin_memcpy(&h, &b, 2);
  unsigned u = ((unsigned)h) << 16;
  float f; __builtin_memcpy(&f, &u, 4); return f;
}
__device__ __forceinline__ float gelu_exact(float x) {
  return 0.5f * x * (1.0f + erff(x * 0.70710678118654752f));
}

#if USE_TDM
// 2D tensor tile -> LDS via TDM. tile = 32 halves x tile1 rows, LDS rows padded
// +16B per 64B (stride-40-halves layout). dims/strides in 2-byte elements.
__device__ __forceinline__ void tdm_load_tile(unsigned ldsAddr, unsigned long long ga,
                                              unsigned tdim0, unsigned tdim1,
                                              unsigned stride, unsigned tile1) {
  v4u g0;
  g0[0] = 1u;                                               // count=1
  g0[1] = ldsAddr;                                          // lds_addr
  g0[2] = (unsigned)(ga & 0xFFFFFFFFu);                     // global_addr lo
  g0[3] = (unsigned)((ga >> 32) & 0x01FFFFFFu) | (2u << 30);// addr[56:32] | type=2
  v8i_ g1;
  g1[0] = (int)((1u << 16)    // data_size = 2B
              | (1u << 20)    // pad_enable
              | (3u << 22)    // pad_interval: every 16 DWORDs
              | (3u << 25));  // pad_amount: 4 DWORDs
  g1[1] = (int)((tdim0 & 0xFFFFu) << 16);
  g1[2] = (int)(((tdim0 >> 16) & 0xFFFFu) | ((tdim1 & 0xFFFFu) << 16));
  g1[3] = (int)(((tdim1 >> 16) & 0xFFFFu) | (32u << 16));   // tile_dim0 = 32
  g1[4] = (int)tile1;                                       // tile_dim1
  g1[5] = (int)stride;                                      // dim0_stride lo32
  g1[6] = 0; g1[7] = 0;
  const v4i_ gz = {0, 0, 0, 0};
  const v8i_ gz8 = {0, 0, 0, 0, 0, 0, 0, 0};
  __builtin_amdgcn_tensor_load_to_lds(g0, g1, gz, gz, gz8, 0);
}
#endif

// ---------- kernel 1: per-segment GroupNorm stats + gate pre-activation ----------
__global__ __launch_bounds__(256)
void seg_stats_kernel(const float* __restrict__ feat, const int* __restrict__ offset,
                      const float* __restrict__ norm_w, const float* __restrict__ norm_b,
                      float* __restrict__ seg_mean, float* __restrict__ seg_rstd,
                      float* __restrict__ g_in) {
  const int b = blockIdx.x;
  const int start = (b == 0) ? 0 : offset[b - 1];
  const int end = offset[b];
  const int cnt = end - start;
  const int t = threadIdx.x;
  float cs0 = 0.f, cs1 = 0.f, s = 0.f, s2 = 0.f;
  for (int r = start; r < end; ++r) {
    const float* row = feat + (size_t)r * CCH;
    float x0 = row[t], x1 = row[t + 256];
    cs0 += x0; cs1 += x1;
    s += x0 + x1; s2 += x0 * x0 + x1 * x1;
  }
  __shared__ float shs[256], shs2[256];
  shs[t] = s; shs2[t] = s2;
  __syncthreads();
  for (int o = 128; o > 0; o >>= 1) {
    if (t < o) { shs[t] += shs[t + o]; shs2[t] += shs2[t + o]; }
    __syncthreads();
  }
  const float denom = (float)cnt * (float)CCH;
  const float mean = shs[0] / denom;
  const float var = shs2[0] / denom - mean * mean;
  const float rstd = rsqrtf(var + EPSF);
  if (t == 0) { seg_mean[b] = mean; seg_rstd[b] = rstd; }
  const float ic = 1.0f / (float)cnt;
  g_in[b * CCH + t]       = (cs0 * ic - mean) * rstd * norm_w[t]       + norm_b[t];
  g_in[b * CCH + t + 256] = (cs1 * ic - mean) * rstd * norm_w[t + 256] + norm_b[t + 256];
}

// ---------- kernel 2: per-row segment id + fused normalization scale/shift ----------
__global__ __launch_bounds__(256)
void row_params_kernel(const int* __restrict__ offset, const float* __restrict__ seg_mean,
                       const float* __restrict__ seg_rstd, int* __restrict__ bid,
                       float* __restrict__ rowScale, float* __restrict__ rowShift,
                       int N, int nseg) {
  const int n = blockIdx.x * 256 + threadIdx.x;
  if (n >= N) return;
  int b = 0;
  while (b < nseg - 1 && n >= offset[b]) ++b;   // searchsorted(side="right")
  const float m = seg_mean[b], r = seg_rstd[b];
  bid[n] = b;
  rowScale[n] = r;
  rowShift[n] = -m * r;
}

// ---------- kernel 3: channel gate (tiny 8x512x512 matmul + sigmoid) ----------
__global__ __launch_bounds__(256)
void gate_kernel(const float* __restrict__ g_in, const float* __restrict__ gate_w,
                 float* __restrict__ gate, int nseg) {
  const int i = blockIdx.x * 256 + threadIdx.x;
  if (i >= nseg * CCH) return;
  const int b = i / CCH, c = i % CCH;
  float acc = 0.f;
  const float* gi = g_in + b * CCH;
  const float* gw = gate_w + (size_t)c * CCH;
  #pragma unroll 8
  for (int cc = 0; cc < CCH; ++cc) acc += gi[cc] * gw[cc];
  gate[i] = 1.0f / (1.0f + expf(-acc));
}

// ---------- prep: f32 -> bf16 weight conversion (once per launch) ----------
__global__ __launch_bounds__(256)
void cvt_bf16_kernel(const float* __restrict__ src, __bf16* __restrict__ dst, int n8) {
  const int i = blockIdx.x * 256 + threadIdx.x;
  if (i >= n8) return;
  const float4 a = *(const float4*)(src + (size_t)i * 8);
  const float4 b = *(const float4*)(src + (size_t)i * 8 + 4);
  v4u u;
  u.x = pkbf(a.x, a.y); u.y = pkbf(a.z, a.w);
  u.z = pkbf(b.x, b.y); u.w = pkbf(b.z, b.w);
  *(v4u*)&dst[(size_t)i * 8] = u;
}

// ---------- prep: combined depthwise kernel 0.5*(fwd[k] + bwd[K-1-k]) ----------
__global__ __launch_bounds__(256)
void conv_w_kernel(const float* __restrict__ fwd_w, const float* __restrict__ bwd_w,
                   float* __restrict__ wc) {
  const int i = blockIdx.x * 256 + threadIdx.x;
  if (i >= HH * KK) return;
  const int h = i / KK, k = i % KK;
  wc[i] = 0.5f * (fwd_w[h * KK + k] + bwd_w[h * KK + (KK - 1 - k)]);
}

// ---------- shared WMMA fragment loads ----------
__device__ __forceinline__ v16bf load_a_frag(const __bf16* As, int tr, int lane) {
  const int am = tr * 16 + (lane & 15);
  const int kb = (lane >> 4) * 8;
  const v8bf lo = *(const v8bf*)&As[am * 40 + kb];
  const v8bf hi = *(const v8bf*)&As[am * 40 + kb + 16];
  v16bf a;
  #pragma unroll
  for (int i = 0; i < 8; ++i) { a[i] = lo[i]; a[i + 8] = hi[i]; }
  return a;
}
__device__ __forceinline__ v16bf load_b_frag(const __bf16* Bs, int bn, int lane) {
  const int n = bn * 16 + (lane & 15);
  const int kb2 = (lane >> 4) * 8;
  const v8bf lo = *(const v8bf*)&Bs[n * 40 + kb2];
  const v8bf hi = *(const v8bf*)&Bs[n * 40 + kb2 + 16];
  v16bf b;
  #pragma unroll
  for (int i = 0; i < 8; ++i) { b[i] = lo[i]; b[i + 8] = hi[i]; }
  return b;
}

// ---------- kernel 4: in_proj GEMM  state[N,H] = gelu( xn[N,C] @ in_proj_w^T ) ----------
// 64x128 block tile, 8 waves, 4 accumulators/wave. Double-buffered LDS:
// A (normalized feat) filled manually (loads before compute, stores after);
// B (bf16 weights) streamed by the Tensor Data Mover, overlapped with WMMAs.
__global__ __launch_bounds__(256)
void inproj_wmma_kernel(const float* __restrict__ feat, const __bf16* __restrict__ wbf,
                        const float* __restrict__ norm_w, const float* __restrict__ norm_b,
                        const float* __restrict__ rowScale, const float* __restrict__ rowShift,
                        __bf16* __restrict__ state, int N) {
  __shared__ __align__(16) __bf16 As[2][64 * 40];
  __shared__ __align__(16) __bf16 Bs[2][128 * 40];
  const int rowBase = blockIdx.x * 64;
  const int colBase = blockIdx.y * 128;
  const int tid = threadIdx.x;
  const int wave = tid >> 5, lane = tid & 31;
  const int tr = wave & 3;
  const int cg = wave >> 2;
  v8f acc[4] = {v8f{}, v8f{}, v8f{}, v8f{}};

  const int arow_i = tid >> 2;
  const int akg = (tid & 3) * 8;
  const int gr = rowBase + arow_i;
  const float vf = (gr < N) ? 1.f : 0.f;
  const int grc = (gr < N) ? gr : 0;
  const float rs = rowScale[grc] * vf;
  const float rsh = rowShift[grc] * vf;
  const float* arow = feat + (size_t)grc * CCH;
#if USE_TDM
  const unsigned long long wBase =
      (unsigned long long)(size_t)(wbf + (size_t)colBase * CCH);
#else
  const __bf16* wrow0 = wbf + (size_t)(colBase + ((tid + 0) >> 2)) * CCH + ((tid + 0) & 3) * 8;
  const __bf16* wrow1 = wbf + (size_t)(colBase + ((tid + 256) >> 2)) * CCH + ((tid + 256) & 3) * 8;
#endif

  // prologue: stage k0 = 0 into buffer 0
  {
    const int k = akg;
    const float4 xa = *(const float4*)(arow + k);
    const float4 xb = *(const float4*)(arow + k + 4);
    const float4 wa = *(const float4*)(norm_w + k);
    const float4 wb = *(const float4*)(norm_w + k + 4);
    const float4 ba = *(const float4*)(norm_b + k);
    const float4 bb = *(const float4*)(norm_b + k + 4);
    v4u u;
    u.x = pkbf((xa.x * rs + rsh) * wa.x + ba.x * vf, (xa.y * rs + rsh) * wa.y + ba.y * vf);
    u.y = pkbf((xa.z * rs + rsh) * wa.z + ba.z * vf, (xa.w * rs + rsh) * wa.w + ba.w * vf);
    u.z = pkbf((xb.x * rs + rsh) * wb.x + bb.x * vf, (xb.y * rs + rsh) * wb.y + bb.y * vf);
    u.w = pkbf((xb.z * rs + rsh) * wb.z + bb.z * vf, (xb.w * rs + rsh) * wb.w + bb.w * vf);
    *(v4u*)&As[0][arow_i * 40 + akg] = u;
  }
#if USE_TDM
  if (wave == 0) {
    tdm_load_tile((unsigned)(size_t)&Bs[0][0], wBase, CCH, HH, CCH, 128);
    __builtin_amdgcn_s_wait_tensorcnt(0);
  }
#else
  {
    v4u u0, u1;
    u0 = *(const v4u*)(wrow0); u1 = *(const v4u*)(wrow1);
    *(v4u*)&Bs[0][((tid + 0) >> 2) * 40 + ((tid + 0) & 3) * 8] = u0;
    *(v4u*)&Bs[0][((tid + 256) >> 2) * 40 + ((tid + 256) & 3) * 8] = u1;
  }
#endif
  __syncthreads();

  int cur = 0;
  for (int k0 = 0; k0 < CCH; k0 += 32) {
    const int nxt = cur ^ 1;
    const bool more = (k0 + 32 < CCH);
    float4 xa, xb, wa, wb, ba, bb;
    if (more) {
      const int k = k0 + 32 + akg;
#if USE_TDM
      if (wave == 0)
        tdm_load_tile((unsigned)(size_t)&Bs[nxt][0],
                      wBase + (unsigned long long)(k0 + 32) * 2ull, CCH, HH, CCH, 128);
#endif
      xa = *(const float4*)(arow + k);
      xb = *(const float4*)(arow + k + 4);
      wa = *(const float4*)(norm_w + k);
      wb = *(const float4*)(norm_w + k + 4);
      ba = *(const float4*)(norm_b + k);
      bb = *(const float4*)(norm_b + k + 4);
    }
    // compute current buffer (overlaps TDM + pending global loads)
    const v16bf a = load_a_frag(As[cur], tr, lane);
    #pragma unroll
    for (int i = 0; i < 4; ++i) {
      const v16bf b = load_b_frag(Bs[cur], cg * 4 + i, lane);
      acc[i] = __builtin_amdgcn_wmma_f32_16x16x32_bf16(false, a, false, b, (short)0,
                                                       acc[i], false, false);
    }
    if (more) {
      v4u u;
      u.x = pkbf((xa.x * rs + rsh) * wa.x + ba.x * vf, (xa.y * rs + rsh) * wa.y + ba.y * vf);
      u.y = pkbf((xa.z * rs + rsh) * wa.z + ba.z * vf, (xa.w * rs + rsh) * wa.w + ba.w * vf);
      u.z = pkbf((xb.x * rs + rsh) * wb.x + bb.x * vf, (xb.y * rs + rsh) * wb.y + bb.y * vf);
      u.w = pkbf((xb.z * rs + rsh) * wb.z + bb.z * vf, (xb.w * rs + rsh) * wb.w + bb.w * vf);
      *(v4u*)&As[nxt][arow_i * 40 + akg] = u;
#if !USE_TDM
      *(v4u*)&Bs[nxt][((tid + 0) >> 2) * 40 + ((tid + 0) & 3) * 8] =
          *(const v4u*)(wrow0 + k0 + 32);
      *(v4u*)&Bs[nxt][((tid + 256) >> 2) * 40 + ((tid + 256) & 3) * 8] =
          *(const v4u*)(wrow1 + k0 + 32);
#endif
    }
#if USE_TDM
    if (wave == 0) __builtin_amdgcn_s_wait_tensorcnt(0);
#endif
    __syncthreads();
    cur = nxt;
  }
  // epilogue
  const int lm = lane & 15, lh = lane >> 4;
  const int rbase = rowBase + tr * 16 + lh * 8;
  #pragma unroll
  for (int i = 0; i < 4; ++i) {
    const int c = colBase + (cg * 4 + i) * 16 + lm;
    #pragma unroll
    for (int r = 0; r < 8; ++r) {
      const int grr = rbase + r;
      if (grr < N) state[(size_t)grr * HH + c] = f2bf(gelu_exact(acc[i][r]));
    }
  }
}

// ---------- kernel 5: bidirectional depthwise conv + gelu (bandwidth-bound) ----------
__global__ __launch_bounds__(256)
void dwconv_kernel(const __bf16* __restrict__ state, const float* __restrict__ wc,
                   const int* __restrict__ bid, __bf16* __restrict__ mixed, int N) {
  const int HV = HH / 8;
  const int tid = blockIdx.x * 256 + threadIdx.x;
  if (tid >= N * HV) return;
  const int n = tid / HV;
  const int h0 = (tid % HV) * 8;
  const int myb = bid[n];
  float acc[8] = {0.f, 0.f, 0.f, 0.f, 0.f, 0.f, 0.f, 0.f};
  #pragma unroll
  for (int k = 0; k < KK; ++k) {
    const int nn = n + k - PADK;
    if (nn < 0 || nn >= N) continue;
    if (bid[nn] != myb) continue;               // segment boundary == zero padding
    const v8bf sv = *(const v8bf*)&state[(size_t)nn * HH + h0];
    #pragma unroll
    for (int j = 0; j < 8; ++j)
      acc[j] += bf2f(sv[j]) * wc[(h0 + j) * KK + k];
  }
  v8bf o;
  #pragma unroll
  for (int j = 0; j < 8; ++j) o[j] = f2bf(gelu_exact(acc[j]));
  *(v8bf*)&mixed[(size_t)n * HH + h0] = o;
}

// ---------- kernel 6: out_proj GEMM + gated residual fusion ----------
// Both A (bf16 mixed) and B (bf16 weights) tiles streamed by TDM, double-buffered.
__global__ __launch_bounds__(256)
void outproj_wmma_kernel(const __bf16* __restrict__ mixed, const __bf16* __restrict__ wbf,
                         const float* __restrict__ feat, const float* __restrict__ gate,
                         const int* __restrict__ bid, const float* __restrict__ fusion_w,
                         float* __restrict__ out, int N) {
  __shared__ __align__(16) __bf16 As[2][64 * 40];
  __shared__ __align__(16) __bf16 Bs[2][128 * 40];
  const int rowBase = blockIdx.x * 64;
  const int colBase = blockIdx.y * 128;
  const int tid = threadIdx.x;
  const int wave = tid >> 5, lane = tid & 31;
  const int tr = wave & 3;
  const int cg = wave >> 2;
  v8f acc[4] = {v8f{}, v8f{}, v8f{}, v8f{}};

#if USE_TDM
  const unsigned long long aBase =
      (unsigned long long)(size_t)(mixed + (size_t)rowBase * HH);
  const unsigned long long bBase =
      (unsigned long long)(size_t)(wbf + (size_t)colBase * HH);
  if (wave == 0) {
    tdm_load_tile((unsigned)(size_t)&As[0][0], aBase, HH, (unsigned)N, HH, 64);
    tdm_load_tile((unsigned)(size_t)&Bs[0][0], bBase, HH, CCH, HH, 128);
    __builtin_amdgcn_s_wait_tensorcnt(0);
  }
  __syncthreads();
#else
  const int arow_i = tid >> 2;
  const int akg = (tid & 3) * 8;
  const int gr = rowBase + arow_i;
  const int grc = (gr < N) ? gr : 0;
  const __bf16* arow = mixed + (size_t)grc * HH;
#endif

  int cur = 0;
  for (int k0 = 0; k0 < HH; k0 += 32) {
    const int nxt = cur ^ 1;
    const bool more = (k0 + 32 < HH);
#if USE_TDM
    if (more && wave == 0) {
      const unsigned long long off = (unsigned long long)(k0 + 32) * 2ull;
      tdm_load_tile((unsigned)(size_t)&As[nxt][0], aBase + off, HH, (unsigned)N, HH, 64);
      tdm_load_tile((unsigned)(size_t)&Bs[nxt][0], bBase + off, HH, CCH, HH, 128);
    }
#else
    if (cur == 0 && k0 == 0) {  // prologue fill (fallback path only)
      *(v4u*)&As[0][arow_i * 40 + akg] = *(const v4u*)(arow + akg);
      *(v4u*)&Bs[0][((tid + 0) >> 2) * 40 + ((tid + 0) & 3) * 8] =
          *(const v4u*)(wbf + (size_t)(colBase + ((tid + 0) >> 2)) * HH + ((tid + 0) & 3) * 8);
      *(v4u*)&Bs[0][((tid + 256) >> 2) * 40 + ((tid + 256) & 3) * 8] =
          *(const v4u*)(wbf + (size_t)(colBase + ((tid + 256) >> 2)) * HH + ((tid + 256) & 3) * 8);
      __syncthreads();
    }
#endif
    const v16bf a = load_a_frag(As[cur], tr, lane);
    #pragma unroll
    for (int i = 0; i < 4; ++i) {
      const v16bf b = load_b_frag(Bs[cur], cg * 4 + i, lane);
      acc[i] = __builtin_amdgcn_wmma_f32_16x16x32_bf16(false, a, false, b, (short)0,
                                                       acc[i], false, false);
    }
#if USE_TDM
    if (wave == 0) __builtin_amdgcn_s_wait_tensorcnt(0);
    __syncthreads();
#else
    __syncthreads();
    if (more) {
      const int kn = k0 + 32;
      *(v4u*)&As[nxt][arow_i * 40 + akg] = *(const v4u*)(arow + kn + akg);
      *(v4u*)&Bs[nxt][((tid + 0) >> 2) * 40 + ((tid + 0) & 3) * 8] =
          *(const v4u*)(wbf + (size_t)(colBase + ((tid + 0) >> 2)) * HH + kn + ((tid + 0) & 3) * 8);
      *(v4u*)&Bs[nxt][((tid + 256) >> 2) * 40 + ((tid + 256) & 3) * 8] =
          *(const v4u*)(wbf + (size_t)(colBase + ((tid + 256) >> 2)) * HH + kn + ((tid + 256) & 3) * 8);
      __syncthreads();
    }
#endif
    cur = nxt;
  }
  const float fw = fusion_w[0];
  const int lm = lane & 15, lh = lane >> 4;
  const int rbase = rowBase + tr * 16 + lh * 8;
  #pragma unroll
  for (int i = 0; i < 4; ++i) {
    const int c = colBase + (cg * 4 + i) * 16 + lm;
    #pragma unroll
    for (int r = 0; r < 8; ++r) {
      const int grr = rbase + r;
      if (grr < N) {
        const float g = gate[bid[grr] * CCH + c];
        out[(size_t)grr * CCH + c] = feat[(size_t)grr * CCH + c] + fw * g * acc[i][r];
      }
    }
  }
}

extern "C" void kernel_launch(void* const* d_in, const int* in_sizes, int n_in,
                              void* d_out, int out_size, void* d_ws, size_t ws_size,
                              hipStream_t stream) {
  const float* feat       = (const float*)d_in[0];
  const int*   offset     = (const int*)d_in[1];
  const float* norm_w     = (const float*)d_in[2];
  const float* norm_b     = (const float*)d_in[3];
  const float* in_proj_w  = (const float*)d_in[4];
  const float* fwd_w      = (const float*)d_in[5];
  const float* bwd_w      = (const float*)d_in[6];
  const float* out_proj_w = (const float*)d_in[7];
  const float* gate_w     = (const float*)d_in[8];
  const float* fusion_w   = (const float*)d_in[9];
  float* out = (float*)d_out;

  const int N = in_sizes[0] / CCH;
  const int nseg = in_sizes[1];

  // workspace layout
  char* ws = (char*)d_ws;
  float* seg_mean = (float*)ws;
  float* seg_rstd = seg_mean + 16;
  float* g_in     = seg_rstd + 16;
  float* gate     = g_in + (size_t)nseg * CCH;
  float* wc       = gate + (size_t)nseg * CCH;           // H*K combined conv weights
  int*   bid      = (int*)(wc + (size_t)HH * KK);
  float* rowScale = (float*)(bid + ((N + 3) & ~3));
  float* rowShift = rowScale + N;
  uintptr_t p = (uintptr_t)(rowShift + N);
  p = (p + 255) & ~(uintptr_t)255;
  __bf16* state   = (__bf16*)p;
  __bf16* mixed   = state + (size_t)N * HH;
  __bf16* wbf_in  = mixed + (size_t)N * HH;              // H x C bf16
  __bf16* wbf_out = wbf_in + (size_t)HH * CCH;           // C x H bf16

  // prep
  cvt_bf16_kernel<<<(HH * CCH / 8 + 255) / 256, 256, 0, stream>>>(in_proj_w, wbf_in,
                                                                  HH * CCH / 8);
  cvt_bf16_kernel<<<(CCH * HH / 8 + 255) / 256, 256, 0, stream>>>(out_proj_w, wbf_out,
                                                                  CCH * HH / 8);
  conv_w_kernel<<<(HH * KK + 255) / 256, 256, 0, stream>>>(fwd_w, bwd_w, wc);

  seg_stats_kernel<<<nseg, 256, 0, stream>>>(feat, offset, norm_w, norm_b,
                                             seg_mean, seg_rstd, g_in);
  row_params_kernel<<<(N + 255) / 256, 256, 0, stream>>>(offset, seg_mean, seg_rstd,
                                                         bid, rowScale, rowShift, N, nseg);
  gate_kernel<<<(nseg * CCH + 255) / 256, 256, 0, stream>>>(g_in, gate_w, gate, nseg);

  dim3 g1((N + 63) / 64, HH / 128);
  inproj_wmma_kernel<<<g1, 256, 0, stream>>>(feat, wbf_in, norm_w, norm_b,
                                             rowScale, rowShift, state, N);

  const int convThreads = N * (HH / 8);
  dwconv_kernel<<<(convThreads + 255) / 256, 256, 0, stream>>>(state, wc, bid, mixed, N);

  dim3 g2((N + 63) / 64, CCH / 128);
  outproj_wmma_kernel<<<g2, 256, 0, stream>>>(mixed, wbf_out, feat, gate, bid,
                                              fusion_w, out, N);
}
